// LaCTSWIGLULayer_86663850099365
// MI455X (gfx1250) — matrix operand order repaired and checked by reading, not compiled, not run on hardware
//
#include <hip/hip_runtime.h>
#include <cstdint>
#include <cstddef>

typedef __bf16 bf16_t;
typedef __attribute__((ext_vector_type(16))) __bf16 v16bf;
typedef __attribute__((ext_vector_type(8)))  float  v8f;
typedef int v4si __attribute__((vector_size(16)));

#define S_LEN   2048
#define D_MODEL 2048
#define H_ATT   16
#define HD      128
#define NHEAD   4
#define FD      512
#define CHUNK_SZ 1024
#define WIN_SZ  1024
#define NB      2
#define M_TOK   (NB * S_LEN)   /* 4096 */
#define BHD     (NB * NHEAD)   /* 8    */
#define BASE_LR (-6.9072552f)  /* 0.001 + log(-expm1(-0.001)) */
#define KT      128            /* GEMM K-tile staged in LDS */

#if defined(__has_builtin)
#if __has_builtin(__builtin_amdgcn_global_load_async_to_lds_b128)
#define ASYNC_G2L 1
#endif
#endif

// ---------------------------------------------------------------------------
// WMMA fragment helpers (CDNA5 16x16x32 bf16 layouts, wave32)
// ---------------------------------------------------------------------------
__device__ __forceinline__ v16bf frag_a_row(const bf16_t* base, int ld, int lane) {
  int row = lane & 15, half = lane >> 4;
  const bf16_t* p  = base + (long)row * ld + half * 16;
  const bf16_t* p2 = p + 8;
  v16bf f;
#pragma unroll
  for (int i = 0; i < 8; ++i) { f[i] = p[i]; f[8 + i] = p2[i]; }
  return f;
}

__device__ __forceinline__ v16bf frag_a_col(const bf16_t* base, int ld, int lane) {
  int row = lane & 15, half = lane >> 4;
  v16bf f;
#pragma unroll
  for (int i = 0; i < 8; ++i) {
    f[i]     = base[(long)(half * 16 + i) * ld + row];
    f[8 + i] = base[(long)(half * 16 + 8 + i) * ld + row];
  }
  return f;
}

__device__ __forceinline__ v16bf frag_b_nt(const bf16_t* base, int ld, int lane) {
  int n = lane & 15, half = lane >> 4;
  const bf16_t* p = base + (long)n * ld + half * 16;
  v16bf f;
#pragma unroll
  for (int i = 0; i < 16; ++i) f[i] = p[i];
  return f;
}

__device__ __forceinline__ v16bf frag_b_nn(const bf16_t* base, int ld, int lane) {
  int n = lane & 15, half = lane >> 4;
  const bf16_t* p = base + (long)(half * 16) * ld + n;
  v16bf f;
#pragma unroll
  for (int i = 0; i < 16; ++i) f[i] = p[(long)i * ld];
  return f;
}

__device__ __forceinline__ v8f wmma_bf16(v16bf a, v16bf b, v8f c) {
  return __builtin_amdgcn_wmma_f32_16x16x32_bf16(false, a, false, b, (short)0, c,
                                                 false, false);
}

// 16-byte global -> LDS copy: async LDS-DMA when the toolchain exposes it,
// else a synchronous register bounce (barriers below make both correct).
// Builtin signature (from hipcc diagnostic): (v4i*, v4i*, imm, imm) with
// generic-address-space int4 pointers; (global_src, lds_dst, offset, cpol).
__device__ __forceinline__ void copy16_g2l(bf16_t* ldsDst, const bf16_t* gSrc) {
#ifdef ASYNC_G2L
  __builtin_amdgcn_global_load_async_to_lds_b128((v4si*)(void*)gSrc,
                                                 (v4si*)(void*)ldsDst, 0, 0);
#else
  *(float4*)(void*)ldsDst = *(const float4*)(const void*)gSrc;
#endif
}

template <int N>
__device__ __forceinline__ void wait_async() {
#ifdef ASYNC_G2L
#if __has_builtin(__builtin_amdgcn_s_wait_asynccnt)
  __builtin_amdgcn_s_wait_asynccnt(N);
#else
  asm volatile("s_wait_asynccnt %0" ::"i"(N) : "memory");
#endif
#endif
}

// ---------------------------------------------------------------------------
// Generic batched WMMA GEMM: C[M,N] (+)= op(A) * op(B)
//   AT=false : A row-major [M,K]    AT=true : A stored [K,M]
//   BNT=true : B stored [N,K]       BNT=false : B stored [K,N]
// block = 128 threads = 4 waves; block tile 128x64 (wave tile 32x64).
// B tile [64 x KT] is staged once per block into LDS (async LDS-DMA,
// double-buffered over K tiles); A fragments batch-loaded per K tile.
// K, M, N must be multiples of KT, 128, 64 respectively (true for all calls).
// ---------------------------------------------------------------------------
template <bool AT, bool BNT>
__global__ __launch_bounds__(128) void gemm_wmma(
    const bf16_t* __restrict__ A, const bf16_t* __restrict__ B,
    float* __restrict__ C, int M, int N, int K, int lda, int ldb, int ldc,
    long aBatch, long bBatch, long cBatch, int accumulate) {
  __shared__ bf16_t bs[2][64 * KT];  // 2 x 16KB
  const int bz = blockIdx.z;
  A += (long)bz * aBatch;
  B += (long)bz * bBatch;
  C += (long)bz * cBatch;
  const int tid = threadIdx.x;
  const int lane = tid & 31;
  const int wave = tid >> 5;
  const int m0 = blockIdx.y * 128 + wave * 32;
  const int n0 = blockIdx.x * 64;
  const int nl = lane & 15, half = lane >> 4;

  // cooperative stage of B tile [n0, n0+64) x [kt, kt+KT): 1024 x 16B chunks
  auto stageB = [&](int buf, int kt) {
#pragma unroll
    for (int c = 0; c < 8; ++c) {
      int chunk = tid + c * 128;
      if (BNT) {  // LDS layout [64 n][KT k]
        int n = chunk >> 4, ko = (chunk & 15) * 8;
        copy16_g2l(&bs[buf][n * KT + ko], B + (long)(n0 + n) * ldb + kt + ko);
      } else {  // LDS layout [KT k][64 n]
        int k = chunk >> 3, no = (chunk & 7) * 8;
        copy16_g2l(&bs[buf][k * 64 + no], B + (long)(kt + k) * ldb + n0 + no);
      }
    }
  };
  auto loadA = [&](int k0, int r) -> v16bf {
    if constexpr (AT)
      return frag_a_col(A + (long)k0 * lda + m0 + r * 16, lda, lane);
    else
      return frag_a_row(A + (long)(m0 + r * 16) * lda + k0, lda, lane);
  };

  v8f acc[2][4];
  if (accumulate) {
#pragma unroll
    for (int r = 0; r < 2; ++r)
#pragma unroll
      for (int j = 0; j < 4; ++j)
#pragma unroll
        for (int v = 0; v < 8; ++v)
          acc[r][j][v] =
              C[(long)(m0 + r * 16 + half * 8 + v) * ldc + n0 + j * 16 + nl];
  } else {
#pragma unroll
    for (int r = 0; r < 2; ++r)
#pragma unroll
      for (int j = 0; j < 4; ++j)
#pragma unroll
        for (int v = 0; v < 8; ++v) acc[r][j][v] = 0.f;
  }

  const int ntiles = K / KT;
  stageB(0, 0);
  for (int i = 0; i < ntiles; ++i) {
    __syncthreads();  // everyone done reading the buffer we stage next
    if (i + 1 < ntiles) {
      stageB((i + 1) & 1, (i + 1) * KT);
      wait_async<8>();  // tile i complete (in-order, 8 newer ops may remain)
    } else {
      wait_async<0>();
    }
    __syncthreads();  // tile i visible to all waves

    // batch-load A fragments for the whole K tile (global latency amortized)
    v16bf afr[4][2];
#pragma unroll
    for (int s = 0; s < 4; ++s)
#pragma unroll
      for (int r = 0; r < 2; ++r) afr[s][r] = loadA(i * KT + s * 32, r);

    const bf16_t* bt = &bs[i & 1][0];
#pragma unroll
    for (int s = 0; s < 4; ++s) {
      v16bf bfr[4];
#pragma unroll
      for (int j = 0; j < 4; ++j) {
        if constexpr (BNT)
          bfr[j] = frag_b_nt(bt + (j * 16) * KT + s * 32, KT, lane);
        else
          bfr[j] = frag_b_nn(bt + (s * 32) * 64 + j * 16, 64, lane);
      }
#pragma unroll
      for (int r = 0; r < 2; ++r)
#pragma unroll
        for (int j = 0; j < 4; ++j)
          acc[r][j] = wmma_bf16(afr[s][r], bfr[j], acc[r][j]);
    }
  }

#pragma unroll
  for (int r = 0; r < 2; ++r)
#pragma unroll
    for (int j = 0; j < 4; ++j)
#pragma unroll
      for (int v = 0; v < 8; ++v)
        C[(long)(m0 + r * 16 + half * 8 + v) * ldc + n0 + j * 16 + nl] =
            acc[r][j][v];
}

// ---------------------------------------------------------------------------
// small helpers / elementwise kernels
// ---------------------------------------------------------------------------
__global__ void cvt_bf16_kernel(const float* __restrict__ in,
                                bf16_t* __restrict__ out, long n) {
  long i = (long)blockIdx.x * blockDim.x + threadIdx.x;
  if (i < n) out[i] = (bf16_t)in[i];
}

__device__ __forceinline__ float block_reduce_add(float v, float* red) {
  int tid = threadIdx.x;
  red[tid] = v;
  __syncthreads();
  for (int s = 128; s > 0; s >>= 1) {
    if (tid < s) red[tid] += red[tid + s];
    __syncthreads();
  }
  float r = red[0];
  __syncthreads();
  return r;
}

// Per-token: full-d rmsnorm(q,k) + RoPE -> aq/ak [b,h,s,hd]; V^T -> avt
// [b,h,hd,s]; silu feature maps + L2 norm -> fq/fk/fv [bh,s,fd]; lr dots ->
// softplus -> l0/l1/l2 [bh,s].
__global__ __launch_bounds__(256) void attn_pre_kernel(
    const float* __restrict__ qkv, const float* __restrict__ hidden,
    const float* __restrict__ q_norm_w, const float* __restrict__ k_norm_w,
    const float* __restrict__ qk_scale, const float* __restrict__ qk_offset,
    const float* __restrict__ lr_w, const float* __restrict__ lr_b,
    bf16_t* __restrict__ aq, bf16_t* __restrict__ ak, bf16_t* __restrict__ avt,
    bf16_t* __restrict__ fq, bf16_t* __restrict__ fk, bf16_t* __restrict__ fv,
    float* __restrict__ l0, float* __restrict__ l1, float* __restrict__ l2) {
  const int t = blockIdx.x;
  const int b = t / S_LEN, si = t % S_LEN;
  const int tid = threadIdx.x;
  __shared__ float red[256];
  __shared__ float qn[D_MODEL];
  __shared__ float kn[D_MODEL];

  const float* qrow = qkv + (long)t * 3 * D_MODEL;
  const float* krow = qrow + D_MODEL;
  const float* vrow = krow + D_MODEL;
  const float* hrow = hidden + (long)t * D_MODEL;

  float qv[8], kv[8], vv[8], hv[8];
  float sq = 0.f, sk = 0.f;
#pragma unroll
  for (int u = 0; u < 8; ++u) {
    int i = tid + u * 256;
    qv[u] = qrow[i]; kv[u] = krow[i]; vv[u] = vrow[i]; hv[u] = hrow[i];
    sq += qv[u] * qv[u]; sk += kv[u] * kv[u];
  }
  float rq = rsqrtf(block_reduce_add(sq, red) / (float)D_MODEL + 1e-6f);
  float rk = rsqrtf(block_reduce_add(sk, red) / (float)D_MODEL + 1e-6f);

  float fqv[8], fkv[8], fvv[8];
  float gq_[4] = {0, 0, 0, 0}, gk_[4] = {0, 0, 0, 0};
#pragma unroll
  for (int u = 0; u < 8; ++u) {
    int i = tid + u * 256;
    float x = qv[u], s0 = 1.f / (1.f + __expf(-x));
    fqv[u] = x * s0 * qk_scale[2 * i + 0] + qk_offset[2 * i + 0];
    x = kv[u]; s0 = 1.f / (1.f + __expf(-x));
    fkv[u] = x * s0 * qk_scale[2 * i + 1] + qk_offset[2 * i + 1];
    x = vv[u]; s0 = 1.f / (1.f + __expf(-x));
    fvv[u] = x * s0;
    int g = i >> 9;
    gq_[g] += fqv[u] * fqv[u];
    gk_[g] += fkv[u] * fkv[u];
  }
  float nq[4], nk[4];
  for (int g = 0; g < 4; ++g) {
    nq[g] = sqrtf(block_reduce_add(gq_[g], red)) + 1e-12f;
    nk[g] = sqrtf(block_reduce_add(gk_[g], red)) + 1e-12f;
  }
  float dots[12];
  for (int j = 0; j < 12; ++j) {
    float d0 = 0.f;
#pragma unroll
    for (int u = 0; u < 8; ++u) {
      int i = tid + u * 256;
      d0 += hv[u] * lr_w[(long)j * D_MODEL + i];
    }
    dots[j] = block_reduce_add(d0, red);
  }
#pragma unroll
  for (int u = 0; u < 8; ++u) {
    int i = tid + u * 256;
    int g = i >> 9, o = i & 511;
    long fidx = ((long)(b * NHEAD + g) * S_LEN + si) * FD + o;
    fq[fidx] = (bf16_t)(fqv[u] / nq[g]);
    fk[fidx] = (bf16_t)(fkv[u] / nk[g]);
    fv[fidx] = (bf16_t)fvv[u];
    int head = i >> 7, f2 = i & 127;
    avt[(((long)b * H_ATT + head) * HD + f2) * S_LEN + si] = (bf16_t)vv[u];
    qn[i] = qv[u] * rq * q_norm_w[i];
    kn[i] = kv[u] * rk * k_norm_w[i];
  }
  __syncthreads();
#pragma unroll
  for (int u = 0; u < 4; ++u) {
    int p = tid + u * 256;  // pair id 0..1023
    int head = p >> 6, f = p & 63;
    int i1 = head * HD + f, i2 = i1 + 64;
    float invf = __powf(500000.f, -((float)(2 * f)) / (float)HD);
    float ang = (float)si * invf;
    float s0, c;
    __sincosf(ang, &s0, &c);
    long base = (((long)b * H_ATT + head) * S_LEN + si) * HD;
    aq[base + f]      = (bf16_t)(qn[i1] * c - qn[i2] * s0);
    aq[base + 64 + f] = (bf16_t)(qn[i2] * c + qn[i1] * s0);
    ak[base + f]      = (bf16_t)(kn[i1] * c - kn[i2] * s0);
    ak[base + 64 + f] = (bf16_t)(kn[i2] * c + kn[i1] * s0);
  }
  if (tid < 12) {
    int which = tid >> 2, nh = tid & 3;
    float x = dots[tid] + lr_b[tid] + BASE_LR;
    float sp = (x > 20.f) ? x : log1pf(__expf(x));
    float* dst = (which == 0) ? l0 : (which == 1) ? l1 : l2;
    dst[((long)b * NHEAD + nh) * S_LEN + si] = sp;
  }
}

// ---------------------------------------------------------------------------
// Windowed flash attention. grid = (s/64, b*H). block = 128 (4 waves),
// each wave owns 16 query rows x full hd=128. S and P*V loops are software
// pipelined with a 4-deep B-fragment ring so WMMAs overlap global loads.
// ---------------------------------------------------------------------------
__device__ __forceinline__ float halfgrp_max(float v) {
#pragma unroll
  for (int m = 1; m < 16; m <<= 1) v = fmaxf(v, __shfl_xor(v, m, 32));
  return v;
}
__device__ __forceinline__ float halfgrp_sum(float v) {
#pragma unroll
  for (int m = 1; m < 16; m <<= 1) v += __shfl_xor(v, m, 32);
  return v;
}

__global__ __launch_bounds__(128) void attn_kernel(
    const bf16_t* __restrict__ aq, const bf16_t* __restrict__ ak,
    const bf16_t* __restrict__ avt, float* __restrict__ attn_out) {
  const int bh = blockIdx.y;
  const int qs = blockIdx.x * 64;
  const int wave = threadIdx.x >> 5;
  const int lane = threadIdx.x & 31;
  const int nl = lane & 15, half = lane >> 4;
  const int mrow0 = qs + wave * 16;
  const float scale = 0.08838834764831845f;  // 1/sqrt(128)

  __shared__ bf16_t pbuf[4][16][64];

  const bf16_t* qbase = aq + ((long)bh * S_LEN + mrow0) * HD;
  v16bf qf[4];
#pragma unroll
  for (int t = 0; t < 4; ++t) qf[t] = frag_a_row(qbase + t * 32, HD, lane);

  v8f oacc[8];
#pragma unroll
  for (int f = 0; f < 8; ++f)
#pragma unroll
    for (int v = 0; v < 8; ++v) oacc[f][v] = 0.f;
  float mcur[8], lsum[8];
#pragma unroll
  for (int v = 0; v < 8; ++v) { mcur[v] = -1e30f; lsum[v] = 0.f; }

  const int ks0 = (qs >= WIN_SZ) ? qs - WIN_SZ : 0;
  for (int ks = ks0; ks <= qs; ks += 64) {
    // ---- S = Q*K^T : 16 (j,t) steps, 4-deep pipelined B ring ----
    v8f sacc[4];
#pragma unroll
    for (int j = 0; j < 4; ++j)
#pragma unroll
      for (int v = 0; v < 8; ++v) sacc[j][v] = 0.f;
    auto loadS = [&](int step) -> v16bf {
      int j = step >> 2, t = step & 3;
      return frag_b_nt(ak + ((long)bh * S_LEN + ks + j * 16) * HD + t * 32, HD,
                       lane);
    };
    v16bf ringS[4];
#pragma unroll
    for (int p = 0; p < 4; ++p) ringS[p] = loadS(p);
#pragma unroll
    for (int step = 0; step < 16; ++step) {
      v16bf nxt;
      if (step + 4 < 16) nxt = loadS(step + 4);
      sacc[step >> 2] = wmma_bf16(qf[step & 3], ringS[step & 3], sacc[step >> 2]);
      if (step + 4 < 16) ringS[step & 3] = nxt;
    }
    // ---- mask + online softmax (scores masked in place in sacc) ----
#pragma unroll
    for (int j = 0; j < 4; ++j)
#pragma unroll
      for (int v = 0; v < 8; ++v) {
        int qrow = mrow0 + half * 8 + v;
        int kcol = ks + j * 16 + nl;
        bool ok = (kcol <= qrow) && (qrow - kcol < WIN_SZ);
        sacc[j][v] = ok ? sacc[j][v] * scale : -1e30f;
      }
    float corr[8];
#pragma unroll
    for (int v = 0; v < 8; ++v) {
      float rm =
          fmaxf(fmaxf(sacc[0][v], sacc[1][v]), fmaxf(sacc[2][v], sacc[3][v]));
      rm = halfgrp_max(rm);
      float newm = fmaxf(mcur[v], rm);
      corr[v] = __expf(mcur[v] - newm);
      mcur[v] = newm;
      lsum[v] *= corr[v];
    }
#pragma unroll
    for (int f = 0; f < 8; ++f)
#pragma unroll
      for (int v = 0; v < 8; ++v) oacc[f][v] *= corr[v];
#pragma unroll
    for (int v = 0; v < 8; ++v) {
      float rs = 0.f;
#pragma unroll
      for (int j = 0; j < 4; ++j) {
        float p = (sacc[j][v] > -1e29f) ? __expf(sacc[j][v] - mcur[v]) : 0.f;
        rs += p;
        pbuf[wave][half * 8 + v][j * 16 + nl] = (bf16_t)p;
      }
      lsum[v] += halfgrp_sum(rs);
    }
    __syncthreads();
    // ---- O += P*V : 16 (t,f) steps, 4-deep pipelined V ring ----
    const bf16_t* pb = &pbuf[wave][0][0];
    v16bf pf[2];
#pragma unroll
    for (int t = 0; t < 2; ++t) pf[t] = frag_a_row(pb + t * 32, 64, lane);
    auto loadV = [&](int step) -> v16bf {
      int t = step >> 3, f = step & 7;
      return frag_b_nt(avt + ((long)bh * HD + f * 16) * S_LEN + ks + t * 32,
                       S_LEN, lane);
    };
    v16bf ringV[4];
#pragma unroll
    for (int p = 0; p < 4; ++p) ringV[p] = loadV(p);
#pragma unroll
    for (int step = 0; step < 16; ++step) {
      v16bf nxt;
      if (step + 4 < 16) nxt = loadV(step + 4);
      oacc[step & 7] = wmma_bf16(pf[step >> 3], ringV[step & 3], oacc[step & 7]);
      if (step + 4 < 16) ringV[step & 3] = nxt;
    }
    __syncthreads();
  }
#pragma unroll
  for (int f = 0; f < 8; ++f)
#pragma unroll
    for (int v = 0; v < 8; ++v) {
      int qrow = mrow0 + half * 8 + v;
      attn_out[((long)bh * S_LEN + qrow) * HD + f * 16 + nl] =
          oacc[f][v] / lsum[v];
    }
}

// ---------------------------------------------------------------------------
// TTT kernels
// ---------------------------------------------------------------------------
__global__ void init_w_kernel(const float* __restrict__ w0,
                              const float* __restrict__ w1,
                              const float* __restrict__ w2, float* w0f,
                              float* w1f, float* w2f, bf16_t* w0b, bf16_t* w1b,
                              bf16_t* w2b) {
  long idx = (long)blockIdx.x * blockDim.x + threadIdx.x;
  const long per = (long)FD * FD;
  if (idx >= (long)BHD * per) return;
  int nh = (int)(idx / per) % NHEAD;
  long r = idx % per;
  float a = w0[(long)nh * per + r]; w0f[idx] = a; w0b[idx] = (bf16_t)a;
  a = w1[(long)nh * per + r];       w1f[idx] = a; w1b[idx] = (bf16_t)a;
  a = w2[(long)nh * per + r];       w2f[idx] = a; w2b[idx] = (bf16_t)a;
}

// backward-of-swiglu elementwise glue: build hid, dgate*l0, dhpre*l2, v*l1
__global__ void ttt_mid_kernel(const float* __restrict__ gate,
                               const float* __restrict__ hpre,
                               const float* __restrict__ dh_,
                               const bf16_t* __restrict__ fv,
                               const float* __restrict__ l0,
                               const float* __restrict__ l1,
                               const float* __restrict__ l2,
                               bf16_t* __restrict__ hid, bf16_t* __restrict__ a0,
                               bf16_t* __restrict__ a2, bf16_t* __restrict__ vl1,
                               int chunk) {
  long idx = (long)blockIdx.x * blockDim.x + threadIdx.x;
  if (idx >= (long)BHD * CHUNK_SZ * FD) return;
  int bh = (int)(idx / ((long)CHUNK_SZ * FD));
  int rem = (int)(idx % ((long)CHUNK_SZ * FD));
  int c = rem / FD, col = rem % FD;
  long tok = (long)bh * S_LEN + chunk * CHUNK_SZ + c;
  float g = gate[idx], hp = hpre[idx], dv = dh_[idx];
  float sig = 1.f / (1.f + __expf(-g));
  float sg = g * sig;
  hid[idx] = (bf16_t)(sg * hp);
  float li0 = l0[tok], li1 = l1[tok], li2 = l2[tok];
  a0[idx] = (bf16_t)(dv * hp * (sig * (1.f + g * (1.f - sig))) * li0);
  a2[idx] = (bf16_t)(dv * sg * li2);
  vl1[idx] = (bf16_t)((float)fv[tok * FD + col] * li1);
}

__global__ void ttt_hq_kernel(const float* __restrict__ gq,
                              const float* __restrict__ hpq,
                              bf16_t* __restrict__ hq) {
  long idx = (long)blockIdx.x * blockDim.x + threadIdx.x;
  if (idx >= (long)BHD * CHUNK_SZ * FD) return;
  float g = gq[idx];
  float sig = 1.f / (1.f + __expf(-g));
  hq[idx] = (bf16_t)(g * sig * hpq[idx]);
}

// rmsnorm(ttt) * w + attn -> bf16 combo, per token
__global__ __launch_bounds__(256) void combine_kernel(
    const float* __restrict__ ttt, const float* __restrict__ attn,
    const float* __restrict__ ttt_norm_w, bf16_t* __restrict__ combo) {
  const int t = blockIdx.x;
  const int b = t / S_LEN, si = t % S_LEN;
  const int tid = threadIdx.x;
  __shared__ float red[256];
  float vals[8];
  float gs[4] = {0, 0, 0, 0};
#pragma unroll
  for (int u = 0; u < 8; ++u) {
    int i = tid + u * 256;
    int nh = i >> 9, o = i & 511;
    float x = ttt[(((long)b * NHEAD + nh) * S_LEN + si) * FD + o];
    vals[u] = x;
    gs[nh] += x * x;
  }
  float rms[4];
  for (int g = 0; g < 4; ++g)
    rms[g] = rsqrtf(block_reduce_add(gs[g], red) / (float)FD + 1e-6f);
#pragma unroll
  for (int u = 0; u < 8; ++u) {
    int i = tid + u * 256;
    int nh = i >> 9, o = i & 511;
    float tn = vals[u] * rms[nh] * ttt_norm_w[o];
    int head = i >> 7, f = i & 127;
    float av = attn[(((long)b * H_ATT + head) * S_LEN + si) * HD + f];
    combo[(long)t * D_MODEL + i] = (bf16_t)(av + tn);
  }
}

// ---------------------------------------------------------------------------
// host side
// ---------------------------------------------------------------------------
extern "C" void kernel_launch(void* const* d_in, const int* in_sizes, int n_in,
                              void* d_out, int out_size, void* d_ws,
                              size_t ws_size, hipStream_t stream) {
  const float* hidden    = (const float*)d_in[0];
  const float* Wqkv      = (const float*)d_in[1];
  const float* Wo        = (const float*)d_in[2];
  const float* q_norm_w  = (const float*)d_in[3];
  const float* k_norm_w  = (const float*)d_in[4];
  const float* w0        = (const float*)d_in[5];
  const float* w1        = (const float*)d_in[6];
  const float* w2        = (const float*)d_in[7];
  const float* lr_w      = (const float*)d_in[8];
  const float* lr_b      = (const float*)d_in[9];
  const float* qk_scale  = (const float*)d_in[10];
  const float* qk_offset = (const float*)d_in[11];
  const float* ttt_nw    = (const float*)d_in[12];
  float* out = (float*)d_out;

  char* ws = (char*)d_ws;
  size_t off = 0;
  auto alloc = [&](size_t bytes) {
    size_t o = off;
    off = (off + bytes + 255) & ~(size_t)255;
    return (void*)(ws + o);
  };
  const long ND = (long)M_TOK * D_MODEL;             // 8.39M
  const long NWF = (long)BHD * FD * FD;              // 2.10M
  const long NCH = (long)BHD * CHUNK_SZ * FD;        // 4.19M

  bf16_t* hidden_bf = (bf16_t*)alloc(ND * 2);
  bf16_t* wqkv_bf   = (bf16_t*)alloc(3 * (long)D_MODEL * D_MODEL * 2);
  bf16_t* wo_bf     = (bf16_t*)alloc((long)D_MODEL * D_MODEL * 2);
  float*  qkv_f     = (float*)alloc(3 * ND * 4);
  bf16_t* aq_bf     = (bf16_t*)alloc(ND * 2);
  bf16_t* ak_bf     = (bf16_t*)alloc(ND * 2);
  bf16_t* avt_bf    = (bf16_t*)alloc(ND * 2);
  float*  attn_f    = (float*)alloc(ND * 4);
  bf16_t* fq_bf     = (bf16_t*)alloc(ND * 2);
  bf16_t* fk_bf     = (bf16_t*)alloc(ND * 2);
  bf16_t* fv_bf     = (bf16_t*)alloc(ND * 2);
  float*  l0_f      = (float*)alloc((long)BHD * S_LEN * 4);
  float*  l1_f      = (float*)alloc((long)BHD * S_LEN * 4);
  float*  l2_f      = (float*)alloc((long)BHD * S_LEN * 4);
  float*  w0_f      = (float*)alloc(NWF * 4);
  float*  w1_f      = (float*)alloc(NWF * 4);
  float*  w2_f      = (float*)alloc(NWF * 4);
  bf16_t* w0_bf     = (bf16_t*)alloc(NWF * 2);
  bf16_t* w1_bf     = (bf16_t*)alloc(NWF * 2);
  bf16_t* w2_bf     = (bf16_t*)alloc(NWF * 2);
  float*  gate_f    = (float*)alloc(NCH * 4);
  float*  hpre_f    = (float*)alloc(NCH * 4);
  float*  dh_f      = (float*)alloc(NCH * 4);
  bf16_t* hid_bf    = (bf16_t*)alloc(NCH * 2);
  bf16_t* a0_bf     = (bf16_t*)alloc(NCH * 2);
  bf16_t* a2_bf     = (bf16_t*)alloc(NCH * 2);
  bf16_t* vl1_bf    = (bf16_t*)alloc(NCH * 2);
  bf16_t* hq_bf     = (bf16_t*)alloc(NCH * 2);
  float*  ttt_f     = (float*)alloc((long)BHD * S_LEN * FD * 4);
  bf16_t* combo_bf  = (bf16_t*)alloc(ND * 2);
  (void)ws_size; (void)in_sizes; (void)n_in; (void)out_size;

  auto blocks = [](long n) { return (unsigned)((n + 255) / 256); };

  // 0) f32 -> bf16 operand conversion
  cvt_bf16_kernel<<<blocks(ND), 256, 0, stream>>>(hidden, hidden_bf, ND);
  cvt_bf16_kernel<<<blocks(3L * D_MODEL * D_MODEL), 256, 0, stream>>>(
      Wqkv, wqkv_bf, 3L * D_MODEL * D_MODEL);
  cvt_bf16_kernel<<<blocks((long)D_MODEL * D_MODEL), 256, 0, stream>>>(
      Wo, wo_bf, (long)D_MODEL * D_MODEL);

  // 1) qkv = hidden @ Wqkv^T   [4096 x 6144]
  gemm_wmma<false, true><<<dim3(3 * D_MODEL / 64, M_TOK / 128, 1), 128, 0,
                           stream>>>(hidden_bf, wqkv_bf, qkv_f, M_TOK,
                                     3 * D_MODEL, D_MODEL, D_MODEL, D_MODEL,
                                     3 * D_MODEL, 0, 0, 0, 0);

  // 2) per-token preprocessing
  attn_pre_kernel<<<M_TOK, 256, 0, stream>>>(
      qkv_f, hidden, q_norm_w, k_norm_w, qk_scale, qk_offset, lr_w, lr_b,
      aq_bf, ak_bf, avt_bf, fq_bf, fk_bf, fv_bf, l0_f, l1_f, l2_f);

  // 3) windowed flash attention
  attn_kernel<<<dim3(S_LEN / 64, NB * H_ATT), 128, 0, stream>>>(
      aq_bf, ak_bf, avt_bf, attn_f);

  // 4) fast-weight init (tile over batch)
  init_w_kernel<<<blocks(NWF), 256, 0, stream>>>(w0, w1, w2, w0_f, w1_f, w2_f,
                                                 w0_bf, w1_bf, w2_bf);

  // 5) TTT scan over 2 chunks
  for (int chunk = 0; chunk < 2; ++chunk) {
    const long co = (long)chunk * CHUNK_SZ * FD;
    dim3 gFwd(FD / 64, CHUNK_SZ / 128, BHD);  // [1024 x 512], K=512
    dim3 gUpd(FD / 64, FD / 128, BHD);        // [512 x 512],  K=1024
    // gate = k @ w0^T ; hpre = k @ w2^T
    gemm_wmma<false, true><<<gFwd, 128, 0, stream>>>(
        fk_bf + co, w0_bf, gate_f, CHUNK_SZ, FD, FD, FD, FD, FD,
        (long)S_LEN * FD, (long)FD * FD, (long)CHUNK_SZ * FD, 0);
    gemm_wmma<false, true><<<gFwd, 128, 0, stream>>>(
        fk_bf + co, w2_bf, hpre_f, CHUNK_SZ, FD, FD, FD, FD, FD,
        (long)S_LEN * FD, (long)FD * FD, (long)CHUNK_SZ * FD, 0);
    // dh = v @ w1 (w1 is [o,h] row-major -> "nn")
    gemm_wmma<false, false><<<gFwd, 128, 0, stream>>>(
        fv_bf + co, w1_bf, dh_f, CHUNK_SZ, FD, FD, FD, FD, FD,
        (long)S_LEN * FD, (long)FD * FD, (long)CHUNK_SZ * FD, 0);
    // elementwise glue
    ttt_mid_kernel<<<blocks(NCH), 256, 0, stream>>>(
        gate_f, hpre_f, dh_f, fv_bf, l0_f, l1_f, l2_f, hid_bf, a0_bf, a2_bf,
        vl1_bf, chunk);
    // fast-weight updates (A^T B outer products, accumulate into f32 masters)
    gemm_wmma<true, false><<<gUpd, 128, 0, stream>>>(
        a0_bf, fk_bf + co, w0_f, FD, FD, CHUNK_SZ, FD, FD, FD,
        (long)CHUNK_SZ * FD, (long)S_LEN * FD, (long)FD * FD, 1);
    gemm_wmma<true, false><<<gUpd, 128, 0, stream>>>(
        a2_bf, fk_bf + co, w2_f, FD, FD, CHUNK_SZ, FD, FD, FD,
        (long)CHUNK_SZ * FD, (long)S_LEN * FD, (long)FD * FD, 1);
    gemm_wmma<true, false><<<gUpd, 128, 0, stream>>>(
        vl1_bf, hid_bf, w1_f, FD, FD, CHUNK_SZ, FD, FD, FD,
        (long)CHUNK_SZ * FD, (long)CHUNK_SZ * FD, (long)FD * FD, 1);
    // refresh bf16 shadows
    cvt_bf16_kernel<<<blocks(NWF), 256, 0, stream>>>(w0_f, w0_bf, NWF);
    cvt_bf16_kernel<<<blocks(NWF), 256, 0, stream>>>(w1_f, w1_bf, NWF);
    cvt_bf16_kernel<<<blocks(NWF), 256, 0, stream>>>(w2_f, w2_bf, NWF);
    // query pass with updated weights (reuse gate/hpre buffers)
    gemm_wmma<false, true><<<gFwd, 128, 0, stream>>>(
        fq_bf + co, w0_bf, gate_f, CHUNK_SZ, FD, FD, FD, FD, FD,
        (long)S_LEN * FD, (long)FD * FD, (long)CHUNK_SZ * FD, 0);
    gemm_wmma<false, true><<<gFwd, 128, 0, stream>>>(
        fq_bf + co, w2_bf, hpre_f, CHUNK_SZ, FD, FD, FD, FD, FD,
        (long)S_LEN * FD, (long)FD * FD, (long)CHUNK_SZ * FD, 0);
    ttt_hq_kernel<<<blocks(NCH), 256, 0, stream>>>(gate_f, hpre_f, hq_bf);
    // o = hq @ w1n^T  (w1 [o,h] row-major is exactly "nt" for this product)
    gemm_wmma<false, true><<<gFwd, 128, 0, stream>>>(
        hq_bf, w1_bf, ttt_f + co, CHUNK_SZ, FD, FD, FD, FD, FD,
        (long)CHUNK_SZ * FD, (long)FD * FD, (long)S_LEN * FD, 0);
  }

  // 6) combine: rmsnorm(ttt)*w + attn -> bf16
  combine_kernel<<<M_TOK, 256, 0, stream>>>(ttt_f, attn_f, ttt_nw, combo_bf);

  // 7) out = combo @ Wo^T
  gemm_wmma<false, true><<<dim3(D_MODEL / 64, M_TOK / 128, 1), 128, 0, stream>>>(
      combo_bf, wo_bf, out, M_TOK, D_MODEL, D_MODEL, D_MODEL, D_MODEL, D_MODEL,
      0, 0, 0, 0);
}
